// HawkBlock_53386443489948
// MI455X (gfx1250) — compile-verified, matrix operand order checked by reading, not run on hardware
//
#include <hip/hip_runtime.h>
#include <math.h>

// ---------------------------------------------------------------------------
// Types for CDNA5 WMMA
// ---------------------------------------------------------------------------
typedef __attribute__((ext_vector_type(16))) __bf16 v16bf;
typedef __attribute__((ext_vector_type(8)))  float  v8f;

#define B_DIM   4
#define S_DIM   4096
#define M_TOT   (B_DIM * S_DIM)   // 16384 tokens
#define D_DIM   1024
#define H_DIM   1536
#define N2      (2 * H_DIM)       // 3072 (paired projections)
#define SC_L    128               // scan chunk length
#define SC_NC   (S_DIM / SC_L)    // 32 chunks

__device__ __forceinline__ unsigned short f2bf(float f) {
    unsigned int u = __builtin_bit_cast(unsigned int, f);
    unsigned int r = u + 0x7FFFu + ((u >> 16) & 1u);   // round-to-nearest-even
    return (unsigned short)(r >> 16);
}

__device__ __forceinline__ float gelu_exact(float x) {
    return 0.5f * x * (1.0f + erff(x * 0.70710678118654752f));
}

__device__ __forceinline__ float sigmoidf(float x) {
    return 1.0f / (1.0f + __expf(-x));
}

// ---------------------------------------------------------------------------
// f32 -> bf16 conversion (weights / packing)
// ---------------------------------------------------------------------------
__global__ void k_f32_to_bf16(const float* __restrict__ src,
                              unsigned short* __restrict__ dst, int n) {
    int i = blockIdx.x * 256 + threadIdx.x;
    if (i < n) dst[i] = f2bf(src[i]);
}

// ---------------------------------------------------------------------------
// RMSNorm: one block per row, output bf16
// ---------------------------------------------------------------------------
__global__ __launch_bounds__(256)
void k_rmsnorm_bf16(const float* __restrict__ x, const float* __restrict__ w,
                    unsigned short* __restrict__ out, int D) {
    int row = blockIdx.x;
    const float* xr = x + (size_t)row * D;
    float s = 0.f;
    for (int i = threadIdx.x; i < D; i += 256) { float v = xr[i]; s += v * v; }
    __shared__ float red[256];
    red[threadIdx.x] = s;
    __syncthreads();
    for (int st = 128; st > 0; st >>= 1) {
        if (threadIdx.x < st) red[threadIdx.x] += red[threadIdx.x + st];
        __syncthreads();
    }
    float scale = rsqrtf(red[0] / (float)D + 1e-5f);
    unsigned short* orow = out + (size_t)row * D;
    for (int i = threadIdx.x; i < D; i += 256)
        orow[i] = f2bf(xr[i] * scale * w[i]);
}

// ---------------------------------------------------------------------------
// Tiled bf16 WMMA GEMM:  C[M,N] = A[M,K] * W[N,K]^T (+bias[N]) (+addend[M,N])
// Block tile 128x128, BK=32, 8 waves of 32 (2x4), wave tile 64x32 (4x2 WMMA).
// ---------------------------------------------------------------------------
__global__ __launch_bounds__(256)
void k_gemm_bf16_wmma(const unsigned short* __restrict__ A,
                      const unsigned short* __restrict__ W,
                      float* __restrict__ C,
                      const float* __restrict__ bias,
                      const float* __restrict__ addend,
                      int M, int N, int K) {
    constexpr int BK = 32;
    constexpr int LDSW = 40;   // row pitch: 80B = 20 banks -> conflict-free b128
    __shared__ __align__(16) unsigned short lA[128 * LDSW];
    __shared__ __align__(16) unsigned short lB[128 * LDSW];

    const int mBase = blockIdx.y * 128;
    const int nBase = blockIdx.x * 128;
    const int t    = threadIdx.x;
    const int wave = t >> 5;
    const int lane = t & 31;
    const int half = (lane >> 4) & 1;   // CDNA5: which K-half this lane holds
    const int l16  = lane & 15;
    const int wm   = wave >> 2;         // 0..1  (64 rows each)
    const int wn   = wave & 3;          // 0..3  (32 cols each)

    const v8f vzero = {0.f, 0.f, 0.f, 0.f, 0.f, 0.f, 0.f, 0.f};
    v8f acc[4][2];
#pragma unroll
    for (int i = 0; i < 4; i++)
#pragma unroll
        for (int j = 0; j < 2; j++) acc[i][j] = vzero;

    union Frag { uint4 q[2]; v16bf v; };

    for (int k0 = 0; k0 < K; k0 += BK) {
        __syncthreads();   // protect LDS from previous iteration's readers
        // Stage 128x32 bf16 tiles of A and W: 512 16B-chunks each, 2 per thread
#pragma unroll
        for (int i = 0; i < 2; i++) {
            int c   = t + i * 256;       // 0..511
            int row = c >> 2;            // 0..127
            int col = (c & 3) << 3;      // 0,8,16,24
            *reinterpret_cast<uint4*>(&lA[row * LDSW + col]) =
                *reinterpret_cast<const uint4*>(A + (size_t)(mBase + row) * K + k0 + col);
            *reinterpret_cast<uint4*>(&lB[row * LDSW + col]) =
                *reinterpret_cast<const uint4*>(W + (size_t)(nBase + row) * K + k0 + col);
        }
        __syncthreads();

        // CDNA5 16-bit fragment layout: lanes 0-15 hold K[0..7]+K[16..23],
        // lanes 16-31 hold K[8..15]+K[24..31]  -> two b128 LDS reads per frag.
        Frag afr[4], bfr[2];
#pragma unroll
        for (int mi = 0; mi < 4; mi++) {
            int r = wm * 64 + mi * 16 + l16;
            afr[mi].q[0] = *reinterpret_cast<const uint4*>(&lA[r * LDSW + half * 8]);
            afr[mi].q[1] = *reinterpret_cast<const uint4*>(&lA[r * LDSW + 16 + half * 8]);
        }
#pragma unroll
        for (int ni = 0; ni < 2; ni++) {
            int r = wn * 32 + ni * 16 + l16;
            bfr[ni].q[0] = *reinterpret_cast<const uint4*>(&lB[r * LDSW + half * 8]);
            bfr[ni].q[1] = *reinterpret_cast<const uint4*>(&lB[r * LDSW + 16 + half * 8]);
        }
#pragma unroll
        for (int mi = 0; mi < 4; mi++)
#pragma unroll
            for (int ni = 0; ni < 2; ni++)
                acc[mi][ni] = __builtin_amdgcn_wmma_f32_16x16x32_bf16(
                    false, afr[mi].v, false, bfr[ni].v,
                    (short)0, acc[mi][ni], false, false);
    }

    // Epilogue: C/D layout -> lane l16 = col, VGPR v = row (v + 8*half)
#pragma unroll
    for (int mi = 0; mi < 4; mi++) {
        int rbase = mBase + wm * 64 + mi * 16 + half * 8;
#pragma unroll
        for (int ni = 0; ni < 2; ni++) {
            int col = nBase + wn * 32 + ni * 16 + l16;
            float bv = bias ? bias[col] : 0.0f;
#pragma unroll
            for (int v = 0; v < 8; v++) {
                size_t off = (size_t)(rbase + v) * N + col;
                float val = acc[mi][ni][v] + bv;
                if (addend) val += addend[off];
                C[off] = val;
            }
        }
    }
}

// ---------------------------------------------------------------------------
// Depthwise causal conv1d (K=4) + bias. u lives in ug[:, 0:H].  Writes f32+bf16.
// ---------------------------------------------------------------------------
__global__ void k_conv(const float* __restrict__ ug, const float* __restrict__ cw,
                       const float* __restrict__ cb,
                       float* __restrict__ ucf, unsigned short* __restrict__ ucb) {
    int h = blockIdx.x * 256 + threadIdx.x;   // < H_DIM
    int m = blockIdx.y;
    int s = m & (S_DIM - 1);
    float acc = cb[h];
#pragma unroll
    for (int k = 0; k < 4; k++) {
        int sp = s - 3 + k;
        if (sp >= 0)
            acc += cw[h * 4 + k] * ug[(size_t)(m - 3 + k) * N2 + h];
    }
    ucf[(size_t)m * H_DIM + h] = acc;
    ucb[(size_t)m * H_DIM + h] = f2bf(acc);
}

// ---------------------------------------------------------------------------
// RG-LRU gate math: igrg[:,0:H]=ig, igrg[:,H:2H]=rg  -> in-place (xbeta, alpha)
// ---------------------------------------------------------------------------
__global__ void k_alpha_xbeta(float* __restrict__ igrg,
                              const float* __restrict__ ucf,
                              const float* __restrict__ lam) {
    int h = blockIdx.x * 256 + threadIdx.x;
    int m = blockIdx.y;
    size_t base = (size_t)m * N2;
    float ig = igrg[base + h];
    float rg = igrg[base + H_DIM + h];
    float sp = logf(1.0f + __expf(lam[h]));            // softplus
    float a  = __expf(-8.0f * sp * sigmoidf(rg));
    float beta = sqrtf(1.0f - a * a + 1e-6f);
    float xb = beta * sigmoidf(ig) * ucf[(size_t)m * H_DIM + h];
    igrg[base + h]          = xb;
    igrg[base + H_DIM + h]  = a;
}

// ---------------------------------------------------------------------------
// 3-phase chunked linear scan:  h_t = a_t*h_{t-1} + xb_t
// Phase 1: per-chunk affine summary (A = prod a, B = folded xb)
// ---------------------------------------------------------------------------
__global__ void k_scan1(const float* __restrict__ igrg,
                        float* __restrict__ cA, float* __restrict__ cB) {
    int h = blockIdx.x * 256 + threadIdx.x;
    int c = blockIdx.y;
    int b = blockIdx.z;
    float Ap = 1.0f, Bs = 0.0f;
    int s0 = c * SC_L;
    for (int s = 0; s < SC_L; s++) {
        size_t row = (size_t)(b * S_DIM + s0 + s);
        float a  = igrg[row * N2 + H_DIM + h];
        float xb = igrg[row * N2 + h];
        Ap = a * Ap;
        Bs = a * Bs + xb;
    }
    size_t idx = ((size_t)(b * SC_NC + c)) * H_DIM + h;
    cA[idx] = Ap;
    cB[idx] = Bs;
}

// Phase 2: serial carry across 32 chunks; stores carry-in into cA (in-place)
__global__ void k_scan2(float* __restrict__ cA, const float* __restrict__ cB) {
    int id = blockIdx.x * 256 + threadIdx.x;
    if (id >= B_DIM * H_DIM) return;
    int b = id / H_DIM, h = id % H_DIM;
    float hp = 0.0f;
    for (int c = 0; c < SC_NC; c++) {
        size_t idx = ((size_t)(b * SC_NC + c)) * H_DIM + h;
        float A = cA[idx], Bv = cB[idx];
        cA[idx] = hp;               // carry-in for this chunk
        hp = A * hp + Bv;
    }
}

// Phase 3: replay chunk with carry-in, write hrec over xbeta slot
__global__ void k_scan3(float* __restrict__ igrg, const float* __restrict__ cA) {
    int h = blockIdx.x * 256 + threadIdx.x;
    int c = blockIdx.y;
    int b = blockIdx.z;
    float hp = cA[((size_t)(b * SC_NC + c)) * H_DIM + h];
    int s0 = c * SC_L;
    for (int s = 0; s < SC_L; s++) {
        size_t row = (size_t)(b * S_DIM + s0 + s);
        float a  = igrg[row * N2 + H_DIM + h];
        float xb = igrg[row * N2 + h];
        hp = a * hp + xb;
        igrg[row * N2 + h] = hp;
    }
}

// ---------------------------------------------------------------------------
// gelu(gate) * hrec -> bf16       (gate in ug[:,H:2H], hrec in igrg[:,0:H])
// ---------------------------------------------------------------------------
__global__ void k_gelu_mul(const float* __restrict__ ug,
                           const float* __restrict__ igrg,
                           unsigned short* __restrict__ outb) {
    int h = blockIdx.x * 256 + threadIdx.x;
    int m = blockIdx.y;
    float g  = ug[(size_t)m * N2 + H_DIM + h];
    float hr = igrg[(size_t)m * N2 + h];
    outb[(size_t)m * H_DIM + h] = f2bf(gelu_exact(g) * hr);
}

// gelu(g) * up -> bf16            (g in gu[:,0:H], up in gu[:,H:2H])
__global__ void k_geglu(const float* __restrict__ gu,
                        unsigned short* __restrict__ outb) {
    int h = blockIdx.x * 256 + threadIdx.x;
    int m = blockIdx.y;
    float g  = gu[(size_t)m * N2 + h];
    float up = gu[(size_t)m * N2 + H_DIM + h];
    outb[(size_t)m * H_DIM + h] = f2bf(gelu_exact(g) * up);
}

// ---------------------------------------------------------------------------
// Launch
// ---------------------------------------------------------------------------
extern "C" void kernel_launch(void* const* d_in, const int* in_sizes, int n_in,
                              void* d_out, int out_size, void* d_ws, size_t ws_size,
                              hipStream_t stream) {
    (void)in_sizes; (void)n_in; (void)out_size; (void)ws_size;

    const float* x          = (const float*)d_in[0];
    const float* W_in       = (const float*)d_in[1];
    const float* W_gate     = (const float*)d_in[2];
    const float* conv_w     = (const float*)d_in[3];
    const float* conv_b     = (const float*)d_in[4];
    const float* W_out      = (const float*)d_in[5];
    const float* rg_in_w    = (const float*)d_in[6];
    const float* rg_in_b    = (const float*)d_in[7];
    const float* rg_gate_w  = (const float*)d_in[8];
    const float* rg_gate_b  = (const float*)d_in[9];
    const float* lam        = (const float*)d_in[10];
    const float* mlp_gate_w = (const float*)d_in[11];
    const float* mlp_up_w   = (const float*)d_in[12];
    const float* mlp_down_w = (const float*)d_in[13];
    const float* n1w        = (const float*)d_in[14];
    const float* n2w        = (const float*)d_in[15];

    char* ws = (char*)d_ws;
    size_t off = 0;
    auto alloc = [&](size_t bytes) -> void* {
        void* p = ws + off;
        off += (bytes + 255) & ~(size_t)255;
        return p;
    };

    unsigned short* wa    = (unsigned short*)alloc((size_t)N2 * D_DIM * 2);      // W_in||W_gate
    unsigned short* wrg   = (unsigned short*)alloc((size_t)N2 * H_DIM * 2);      // rg_in||rg_gate
    unsigned short* wout  = (unsigned short*)alloc((size_t)D_DIM * H_DIM * 2);
    unsigned short* wmlp  = (unsigned short*)alloc((size_t)N2 * D_DIM * 2);      // mlp_gate||mlp_up
    unsigned short* wdown = (unsigned short*)alloc((size_t)D_DIM * H_DIM * 2);
    float*          rgb   = (float*)alloc((size_t)N2 * 4);                       // rg bias concat
    unsigned short* h_bf  = (unsigned short*)alloc((size_t)M_TOT * D_DIM * 2);   // norm out (bf16)
    float*          ug    = (float*)alloc((size_t)M_TOT * N2 * 4);               // u||gate, later g||up
    float*          ucf   = (float*)alloc((size_t)M_TOT * H_DIM * 4);            // conv out f32
    unsigned short* ucb   = (unsigned short*)alloc((size_t)M_TOT * H_DIM * 2);   // bf16 act (reused)
    float*          igrg  = (float*)alloc((size_t)M_TOT * N2 * 4);               // ig||rg -> xb||a -> h
    float*          x2    = (float*)alloc((size_t)M_TOT * D_DIM * 4);            // x + mix
    float*          cA    = (float*)alloc((size_t)B_DIM * SC_NC * H_DIM * 4);
    float*          cB    = (float*)alloc((size_t)B_DIM * SC_NC * H_DIM * 4);

    auto cvt = [&](const float* s, unsigned short* d, int n) {
        k_f32_to_bf16<<<(n + 255) / 256, 256, 0, stream>>>(s, d, n);
    };

    // --- weight conversion / packing ---
    cvt(W_in,       wa,                     H_DIM * D_DIM);
    cvt(W_gate,     wa + (size_t)H_DIM * D_DIM, H_DIM * D_DIM);
    cvt(rg_in_w,    wrg,                    H_DIM * H_DIM);
    cvt(rg_gate_w,  wrg + (size_t)H_DIM * H_DIM, H_DIM * H_DIM);
    cvt(W_out,      wout,                   D_DIM * H_DIM);
    cvt(mlp_gate_w, wmlp,                   H_DIM * D_DIM);
    cvt(mlp_up_w,   wmlp + (size_t)H_DIM * D_DIM, H_DIM * D_DIM);
    cvt(mlp_down_w, wdown,                  D_DIM * H_DIM);
    hipMemcpyAsync(rgb,          rg_in_b,   H_DIM * 4, hipMemcpyDeviceToDevice, stream);
    hipMemcpyAsync(rgb + H_DIM,  rg_gate_b, H_DIM * 4, hipMemcpyDeviceToDevice, stream);

    dim3 blk(256);
    dim3 gEw(H_DIM / 256, M_TOT);          // (6, 16384) elementwise grids

    // --- Hawk mixer branch ---
    k_rmsnorm_bf16<<<M_TOT, blk, 0, stream>>>(x, n1w, h_bf, D_DIM);

    k_gemm_bf16_wmma<<<dim3(N2 / 128, M_TOT / 128), blk, 0, stream>>>(
        h_bf, wa, ug, nullptr, nullptr, M_TOT, N2, D_DIM);        // u || gate

    k_conv<<<gEw, blk, 0, stream>>>(ug, conv_w, conv_b, ucf, ucb);

    k_gemm_bf16_wmma<<<dim3(N2 / 128, M_TOT / 128), blk, 0, stream>>>(
        ucb, wrg, igrg, rgb, nullptr, M_TOT, N2, H_DIM);          // ig || rg

    k_alpha_xbeta<<<gEw, blk, 0, stream>>>(igrg, ucf, lam);
    k_scan1<<<dim3(H_DIM / 256, SC_NC, B_DIM), blk, 0, stream>>>(igrg, cA, cB);
    k_scan2<<<(B_DIM * H_DIM + 255) / 256, blk, 0, stream>>>(cA, cB);
    k_scan3<<<dim3(H_DIM / 256, SC_NC, B_DIM), blk, 0, stream>>>(igrg, cA);

    k_gelu_mul<<<gEw, blk, 0, stream>>>(ug, igrg, ucb);           // gelu(gate)*hrec

    k_gemm_bf16_wmma<<<dim3(D_DIM / 128, M_TOT / 128), blk, 0, stream>>>(
        ucb, wout, x2, nullptr, x, M_TOT, D_DIM, H_DIM);          // x + mix

    // --- GEGLU MLP branch ---
    k_rmsnorm_bf16<<<M_TOT, blk, 0, stream>>>(x2, n2w, h_bf, D_DIM);

    k_gemm_bf16_wmma<<<dim3(N2 / 128, M_TOT / 128), blk, 0, stream>>>(
        h_bf, wmlp, ug, nullptr, nullptr, M_TOT, N2, D_DIM);      // g || up

    k_geglu<<<gEw, blk, 0, stream>>>(ug, ucb);

    k_gemm_bf16_wmma<<<dim3(D_DIM / 128, M_TOT / 128), blk, 0, stream>>>(
        ucb, wdown, (float*)d_out, nullptr, x2, M_TOT, D_DIM, H_DIM);  // x2 + mlp
}
